// GCN_36283883717325
// MI455X (gfx1250) — matrix operand order repaired and checked by reading, not compiled, not run on hardware
//
#include <hip/hip_runtime.h>

#define NFEAT  512
#define NHID   64
#define NCLASS 40

typedef __attribute__((ext_vector_type(16))) __bf16 v16bf;
typedef __attribute__((ext_vector_type(8)))  float  v8f;

union AFrag { v16bf v; unsigned u[8]; };

// f32 -> bf16 pair via native conversion (lets clang emit packed cvt hardware op)
__device__ __forceinline__ unsigned pack_bf16x2(float lo, float hi) {
  union { __bf16 h[2]; unsigned u; } p;
  p.h[0] = (__bf16)lo;
  p.h[1] = (__bf16)hi;
  return p.u;
}

// ---------------------------------------------------------------------------
// GEMM1: xW1[N,64] = x[N,512] @ W1[512,64]   (bf16 WMMA, f32 accum)
// block: 256 threads = 8 waves, 16 rows/wave -> 128 rows/block
// ---------------------------------------------------------------------------
__global__ __launch_bounds__(256)
void gcn_gemm1(const float* __restrict__ x, const float* __restrict__ W1,
               float* __restrict__ xW1, int N) {
  // B fragments of W1 in WMMA layout: [ks(16)][ctile(4)][lane(32)][8 dwords] = 64 KB
  __shared__ unsigned ldsB[16 * 4 * 32 * 8];
  const int tid = threadIdx.x;

  for (int i = tid; i < 16 * 4 * 32 * 8; i += 256) {
    int v    = i & 7;
    int lane = (i >> 3) & 31;
    int c    = (i >> 8) & 3;
    int ks   = i >> 10;
    int k    = ks * 32 + (v < 4 ? 2 * v : 16 + 2 * (v - 4)) + (lane >> 4) * 8;   // B layout: lanes 0-15 K {0..7,16..23}, 16-31 +8
    int col  = c * 16 + (lane & 15);
    ldsB[i] = pack_bf16x2(W1[(size_t)k * NHID + col], W1[(size_t)(k + 1) * NHID + col]);
  }
  __syncthreads();

  const int wave = tid >> 5;
  const int lane = tid & 31;
  const int m    = lane & 15;
  const int hs   = lane >> 4;

  const int row0 = blockIdx.x * 128 + wave * 16;
  int rowA = row0 + m;
  if (rowA > N - 1) rowA = N - 1;                 // clamp for padded loads
  const float* xr = x + (size_t)rowA * NFEAT;

  v8f acc0 = {}, acc1 = {}, acc2 = {}, acc3 = {};

  for (int ks = 0; ks < 16; ++ks) {
    AFrag a;
#pragma unroll
    for (int v = 0; v < 8; ++v) {
      // ISA 16-bit A layout: lanes 0-15 -> K {0..7,16..23}, lanes 16-31 -> +8
      int k = ks * 32 + (v < 4 ? 2 * v : 16 + 2 * (v - 4)) + hs * 8;
      float2 f = *(const float2*)(xr + k);
      a.u[v] = pack_bf16x2(f.x, f.y);
    }
    if (ks < 15) __builtin_prefetch(xr + (ks + 1) * 32 + hs * 8, 0, 1);

    const unsigned* bb = &ldsB[(ks * 4) * 32 * 8 + lane * 8];
    const v16bf b0 = *(const v16bf*)(bb + 0 * 32 * 8);
    const v16bf b1 = *(const v16bf*)(bb + 1 * 32 * 8);
    const v16bf b2 = *(const v16bf*)(bb + 2 * 32 * 8);
    const v16bf b3 = *(const v16bf*)(bb + 3 * 32 * 8);

    acc0 = __builtin_amdgcn_wmma_f32_16x16x32_bf16(false, a.v, false, b0, (short)0, acc0, false, false);
    acc1 = __builtin_amdgcn_wmma_f32_16x16x32_bf16(false, a.v, false, b1, (short)0, acc1, false, false);
    acc2 = __builtin_amdgcn_wmma_f32_16x16x32_bf16(false, a.v, false, b2, (short)0, acc2, false, false);
    acc3 = __builtin_amdgcn_wmma_f32_16x16x32_bf16(false, a.v, false, b3, (short)0, acc3, false, false);
  }

  // D layout: lane L, VGPR r -> M = r + 8*(L>=16), N = L&15
  const int rbase = row0 + hs * 8;
#pragma unroll
  for (int r = 0; r < 8; ++r) {
    int row = rbase + r;
    if (row < N) {
      float* o = xW1 + (size_t)row * NHID + m;
      o[0]  = acc0[r];
      o[16] = acc1[r];
      o[32] = acc2[r];
      o[48] = acc3[r];
    }
  }
}

// ---------------------------------------------------------------------------
// GEMM2: hW2[N,40] = relu(s + b1) @ W2[64,40]   (cols padded to 48 = 3 tiles)
// ---------------------------------------------------------------------------
__global__ __launch_bounds__(256)
void gcn_gemm2(const float* __restrict__ s, const float* __restrict__ b1,
               const float* __restrict__ W2, float* __restrict__ hW2, int N) {
  __shared__ unsigned ldsB[2 * 3 * 32 * 8];   // 6 KB
  __shared__ float b1s[NHID];
  const int tid = threadIdx.x;

  if (tid < NHID) b1s[tid] = b1[tid];
  for (int i = tid; i < 2 * 3 * 32 * 8; i += 256) {
    int v    = i & 7;
    int lane = (i >> 3) & 31;
    int rem  = i >> 8;
    int c    = rem % 3;
    int ks   = rem / 3;
    int k    = ks * 32 + (v < 4 ? 2 * v : 16 + 2 * (v - 4)) + (lane >> 4) * 8;
    int col  = c * 16 + (lane & 15);
    float f0 = (col < NCLASS) ? W2[(size_t)k * NCLASS + col] : 0.0f;
    float f1 = (col < NCLASS) ? W2[(size_t)(k + 1) * NCLASS + col] : 0.0f;
    ldsB[i] = pack_bf16x2(f0, f1);
  }
  __syncthreads();

  const int wave = tid >> 5;
  const int lane = tid & 31;
  const int m    = lane & 15;
  const int hs   = lane >> 4;

  const int row0 = blockIdx.x * 128 + wave * 16;
  int rowA = row0 + m;
  if (rowA > N - 1) rowA = N - 1;
  const float* sr = s + (size_t)rowA * NHID;

  v8f acc0 = {}, acc1 = {}, acc2 = {};

#pragma unroll
  for (int ks = 0; ks < 2; ++ks) {
    AFrag a;
#pragma unroll
    for (int v = 0; v < 8; ++v) {
      int k = ks * 32 + (v < 4 ? 2 * v : 16 + 2 * (v - 4)) + hs * 8;
      float2 f = *(const float2*)(sr + k);
      float h0 = fmaxf(f.x + b1s[k],     0.0f);   // fused bias + relu
      float h1 = fmaxf(f.y + b1s[k + 1], 0.0f);
      a.u[v] = pack_bf16x2(h0, h1);
    }
    const unsigned* bb = &ldsB[(ks * 3) * 32 * 8 + lane * 8];
    const v16bf b0  = *(const v16bf*)(bb + 0 * 32 * 8);
    const v16bf b1v = *(const v16bf*)(bb + 1 * 32 * 8);
    const v16bf b2v = *(const v16bf*)(bb + 2 * 32 * 8);

    acc0 = __builtin_amdgcn_wmma_f32_16x16x32_bf16(false, a.v, false, b0,  (short)0, acc0, false, false);
    acc1 = __builtin_amdgcn_wmma_f32_16x16x32_bf16(false, a.v, false, b1v, (short)0, acc1, false, false);
    acc2 = __builtin_amdgcn_wmma_f32_16x16x32_bf16(false, a.v, false, b2v, (short)0, acc2, false, false);
  }

  const int rbase = row0 + hs * 8;
#pragma unroll
  for (int r = 0; r < 8; ++r) {
    int row = rbase + r;
    if (row < N) {
      float* o = hW2 + (size_t)row * NCLASS;
      o[m]      = acc0[r];
      o[m + 16] = acc1[r];
      if (m < 8) o[m + 32] = acc2[r];   // cols 40..47 are padding
    }
  }
}

// ---------------------------------------------------------------------------
// SpMM (64 feats): dst[row] += val * src[col],  16 lanes/edge, float4 each
// ---------------------------------------------------------------------------
__global__ void gcn_spmm64(const int* __restrict__ row, const int* __restrict__ col,
                           const float* __restrict__ val, const float* __restrict__ src,
                           float* __restrict__ dst, int E) {
  long long tid = (long long)blockIdx.x * blockDim.x + threadIdx.x;
  if (tid >= (long long)E * 16) return;
  int e  = (int)(tid >> 4);
  int fb = ((int)tid & 15) * 4;
  int r = row[e], c = col[e];
  float v = val[e];
  const float4 xv = *(const float4*)(src + (size_t)c * 64 + fb);
  float* d = dst + (size_t)r * 64 + fb;
  unsafeAtomicAdd(d + 0, v * xv.x);
  unsafeAtomicAdd(d + 1, v * xv.y);
  unsafeAtomicAdd(d + 2, v * xv.z);
  unsafeAtomicAdd(d + 3, v * xv.w);
}

// SpMM (40 feats): one atomic per (edge, feature)
__global__ void gcn_spmm40(const int* __restrict__ row, const int* __restrict__ col,
                           const float* __restrict__ val, const float* __restrict__ src,
                           float* __restrict__ dst, int E) {
  unsigned tid = blockIdx.x * blockDim.x + threadIdx.x;
  if (tid >= (unsigned)E * 40u) return;
  unsigned e = tid / 40u;
  unsigned f = tid - e * 40u;
  unsafeAtomicAdd(dst + (size_t)row[e] * NCLASS + f,
                  val[e] * src[(size_t)col[e] * NCLASS + f]);
}

// SpMM (40 feats) with fused +b2 on the gathered source (final pvt spmm)
__global__ void gcn_spmm40b(const int* __restrict__ row, const int* __restrict__ col,
                            const float* __restrict__ val, const float* __restrict__ src,
                            const float* __restrict__ b2, float* __restrict__ dst, int E) {
  unsigned tid = blockIdx.x * blockDim.x + threadIdx.x;
  if (tid >= (unsigned)E * 40u) return;
  unsigned e = tid / 40u;
  unsigned f = tid - e * 40u;
  unsafeAtomicAdd(dst + (size_t)row[e] * NCLASS + f,
                  val[e] * (src[(size_t)col[e] * NCLASS + f] + b2[f]));
}

// ---------------------------------------------------------------------------
// log_softmax over 40 classes, one wave32 per row, lane covers f and f+32
// ---------------------------------------------------------------------------
__global__ void gcn_logsoftmax(float* __restrict__ z, int nrows) {
  int wv   = (int)((blockIdx.x * blockDim.x + threadIdx.x) >> 5);
  int lane = threadIdx.x & 31;
  if (wv >= nrows) return;
  float* r = z + (size_t)wv * NCLASS;
  float v0 = r[lane];
  float v1 = (lane < 8) ? r[lane + 32] : -3.0e38f;
  float mx = fmaxf(v0, v1);
#pragma unroll
  for (int o = 16; o > 0; o >>= 1) mx = fmaxf(mx, __shfl_xor(mx, o, 32));
  float se = __expf(v0 - mx) + ((lane < 8) ? __expf(v1 - mx) : 0.0f);
#pragma unroll
  for (int o = 16; o > 0; o >>= 1) se += __shfl_xor(se, o, 32);
  float lse = mx + __logf(se);
  r[lane] = v0 - lse;
  if (lane < 8) r[lane + 32] = v1 - lse;
}

__global__ void gcn_zero(float* __restrict__ p, long long n) {
  long long i = (long long)blockIdx.x * blockDim.x + threadIdx.x;
  if (i < n) p[i] = 0.0f;
}

// ---------------------------------------------------------------------------
extern "C" void kernel_launch(void* const* d_in, const int* in_sizes, int n_in,
                              void* d_out, int out_size, void* d_ws, size_t ws_size,
                              hipStream_t stream) {
  const float* x       = (const float*)d_in[0];
  const int*   adj_row = (const int*)  d_in[1];
  const int*   adj_col = (const int*)  d_in[2];
  const float* adj_val = (const float*)d_in[3];
  const int*   pvt_row = (const int*)  d_in[4];
  const int*   pvt_col = (const int*)  d_in[5];
  const float* pvt_val = (const float*)d_in[6];
  const float* W1      = (const float*)d_in[7];
  const float* b1      = (const float*)d_in[8];
  const float* W2      = (const float*)d_in[9];
  const float* b2      = (const float*)d_in[10];

  const int N  = in_sizes[0] / NFEAT;
  const int E  = in_sizes[1];
  const int E2 = in_sizes[4];

  float* out = (float*)d_out;
  float* xW1 = (float*)d_ws;                       // [N,64]
  float* s   = xW1 + (size_t)N * NHID;             // [N,64]
  float* hW2 = s   + (size_t)N * NHID;             // [N,40]
  float* t   = hW2 + (size_t)N * NCLASS;           // [N,40]

  const long long ns = (long long)N * NHID;
  const long long nt = (long long)N * NCLASS;

  gcn_zero<<<(unsigned)((ns + 255) / 256), 256, 0, stream>>>(s, ns);
  gcn_zero<<<(unsigned)((nt + 255) / 256), 256, 0, stream>>>(t, nt);
  gcn_zero<<<(unsigned)((nt + 255) / 256), 256, 0, stream>>>(out, nt);

  const unsigned gemmBlocks = (unsigned)((N + 127) / 128);
  gcn_gemm1<<<gemmBlocks, 256, 0, stream>>>(x, W1, xW1, N);

  long long w1 = (long long)E * 16;
  gcn_spmm64<<<(unsigned)((w1 + 255) / 256), 256, 0, stream>>>(adj_row, adj_col, adj_val, xW1, s, E);

  gcn_gemm2<<<gemmBlocks, 256, 0, stream>>>(s, b1, W2, hW2, N);

  long long w2 = (long long)E * 40;
  gcn_spmm40<<<(unsigned)((w2 + 255) / 256), 256, 0, stream>>>(adj_row, adj_col, adj_val, hW2, t, E);

  long long w3 = (long long)E2 * 40;
  gcn_spmm40b<<<(unsigned)((w3 + 255) / 256), 256, 0, stream>>>(pvt_row, pvt_col, pvt_val, t, b2, out, E2);

  long long wlsm = (long long)N * 32;
  gcn_logsoftmax<<<(unsigned)((wlsm + 255) / 256), 256, 0, stream>>>(out, N);
}
